// Codebook_824633721588
// MI455X (gfx1250) — matrix operand order repaired and checked by reading, not compiled, not run on hardware
//
#include <hip/hip_runtime.h>

typedef __attribute__((ext_vector_type(16))) __bf16 v16bf;
typedef __attribute__((ext_vector_type(8)))  float  v8f;

#define D_LAT   256
#define K_CODES 8192
#define N_ROWS  32768      // 32*32*32 latent vectors
#define NUM_EL  8388608    // 32*256*32*32
#define BETA    0.25f
#define NTILES  (K_CODES / 16)
#define TPITCH  264        // bf16 pitch per code row in LDS (528B = 33*16B)

#if defined(__has_builtin)
#if __has_builtin(__builtin_amdgcn_global_load_async_to_lds_b128) && \
    __has_builtin(__builtin_amdgcn_s_wait_asynccnt)
#define USE_ASYNC_LDS 1
#endif
#endif

union AFrag {
  unsigned short us[16];
  uint4 q[2];
  v16bf v;
};
static_assert(sizeof(AFrag) == 32, "frag must be 8 VGPRs");

#if defined(USE_ASYNC_LDS)
typedef __attribute__((vector_size(16))) int v4i_t;
typedef __attribute__((address_space(1))) v4i_t* g_v4i;
typedef __attribute__((address_space(3))) v4i_t* l_v4i;

__device__ __forceinline__ void cp16_async(const void* g, void* l) {
  // GLOBAL_LOAD_ASYNC_TO_LDS_B128: per-lane 16B memory -> LDS, ASYNCcnt-tracked.
  // Global VA == generic VA; LDS offset == low 32 bits of the generic address
  // (ISA aperture rule: generic LDS addr = {aperture[63:32], offset[31:0]}).
  __builtin_amdgcn_global_load_async_to_lds_b128(
      (g_v4i)(unsigned long long)g,
      (l_v4i)(unsigned int)(unsigned long long)l, 0, 0);
}
#endif

// ---------------------------------------------------------------------------
// Kernel 1: split codebook fp32 -> bf16 hi/lo planes + per-code squared norm.
// One wave (32 lanes) per code row; fully coalesced 128B reads.
// ---------------------------------------------------------------------------
__global__ __launch_bounds__(256) void vq_prep(
    const float* __restrict__ emb,
    unsigned short* __restrict__ ehi,
    unsigned short* __restrict__ elo,
    float* __restrict__ enorm)
{
  const int t = threadIdx.x, wave = t >> 5, lane = t & 31;
  const int code = blockIdx.x * 8 + wave;
  const float* row = emb + code * D_LAT;
  float ss = 0.f;
#pragma unroll
  for (int k = 0; k < 8; ++k) {
    const int d = lane + 32 * k;
    const float f = row[d];
    ss += f * f;
    const unsigned fb = __float_as_uint(f);
    const float fh = __uint_as_float(fb & 0xFFFF0000u);   // exact hi part
    const float fl = f - fh;                              // exact residual
    ehi[code * D_LAT + d] = (unsigned short)(fb >> 16);
    elo[code * D_LAT + d] = (unsigned short)(__float_as_uint(fl) >> 16);
  }
#pragma unroll
  for (int mask = 1; mask < 32; mask <<= 1) ss += __shfl_xor(ss, mask, 32);
  if (lane == 0) enorm[code] = ss;
}

// ---------------------------------------------------------------------------
// Kernel 2: fused distance GEMM + argmin, double-buffered LDS pipeline.
// Each wave owns 16 z-rows (A tile in registers, split bf16 hi/lo).
// Per 16-code tile: 24 x v_wmma_f32_16x16x32_bf16
//   acc += Ahi*Bhi + Ahi*Blo + Alo*Bhi   (fp32-class accuracy from bf16 units)
// Tile cb+1 streams into LDS (async DMA when available) while tile cb computes.
// ---------------------------------------------------------------------------
__global__ __launch_bounds__(256) void vq_argmin(
    const float* __restrict__ z,
    const unsigned short* __restrict__ ehi,
    const unsigned short* __restrict__ elo,
    const float* __restrict__ enorm,
    int* __restrict__ idx_ws,
    float* __restrict__ idx_out)
{
  // pitch 264 bf16 (132 dwords): aligned b128 reads, no 16-lane bank conflicts
  __shared__ __align__(16) unsigned short sh_hi[2][16 * TPITCH];
  __shared__ __align__(16) unsigned short sh_lo[2][16 * TPITCH];

  const int t    = threadIdx.x;
  const int wave = t >> 5;
  const int lane = t & 31;
  const int l    = lane & 15;    // A row / B column / C column
  const int hi   = lane >> 4;    // half select per ISA fragment layout
  const int m0   = (blockIdx.x * 8 + wave) * 16;

  // Per-thread copy slice: 2 x 16B per plane per tile (512 uint4 per plane).
  const int e0 = t,       c0 = e0 >> 5, o0 = c0 * TPITCH + (e0 & 31) * 8;
  const int e1 = t + 256, c1 = e1 >> 5, o1 = c1 * TPITCH + (e1 & 31) * 8;

#if defined(USE_ASYNC_LDS)
#define TILE_CP(src, dst) cp16_async((const void*)(src), (void*)(dst))
#else
#define TILE_CP(src, dst) (*(uint4*)(dst) = *(const uint4*)(src))
#endif

#define COPY_TILE(cbi, buf)                                                   \
  do {                                                                        \
    const uint4* _sh = (const uint4*)(ehi + (cbi) * (16 * D_LAT));            \
    const uint4* _sl = (const uint4*)(elo + (cbi) * (16 * D_LAT));            \
    TILE_CP(&_sh[e0], &sh_hi[(buf)][o0]);                                     \
    TILE_CP(&_sl[e0], &sh_lo[(buf)][o0]);                                     \
    TILE_CP(&_sh[e1], &sh_hi[(buf)][o1]);                                     \
    TILE_CP(&_sl[e1], &sh_lo[(buf)][o1]);                                     \
  } while (0)

  // prologue: start DMA of tile 0 while we gather the A fragments
  COPY_TILE(0, 0);

  // --- build A fragments from z (NCHW, so d has stride H*W = 1024 floats) ---
  const int n = m0 + l;
  const int b = n >> 10, h = (n >> 5) & 31, w = n & 31;
  const int zbase = b * 262144 + h * 32 + w;   // + d*1024

  AFrag a_hi[8], a_lo[8];
#pragma unroll
  for (int kc = 0; kc < 8; ++kc) {
#pragma unroll
    for (int j = 0; j < 16; ++j) {
      // A k order chosen here (same as B): lane (l,hi) holds K = 16*hi + j
      const int d = 32 * kc + 16 * hi + j;
      const float f = z[zbase + d * 1024];
      const unsigned fb = __float_as_uint(f);
      const float fh = __uint_as_float(fb & 0xFFFF0000u);
      const float fl = f - fh;
      a_hi[kc].us[j] = (unsigned short)(fb >> 16);
      a_lo[kc].us[j] = (unsigned short)(__float_as_uint(fl) >> 16);
    }
  }

  float minval[8];
  int   minidx[8];
#pragma unroll
  for (int r = 0; r < 8; ++r) { minval[r] = 3.4e38f; minidx[r] = 0; }

  for (int cb = 0; cb < NTILES; ++cb) {
    const int cur = cb & 1;
    // kick off next tile into the other buffer (its readers finished at the
    // trailing barrier of iteration cb-1), then wait for the current tile.
    if (cb + 1 < NTILES) {
      COPY_TILE(cb + 1, cur ^ 1);
#if defined(USE_ASYNC_LDS)
      __builtin_amdgcn_s_wait_asynccnt(4);   // in-order: tile cb's 4 ops done
#endif
    } else {
#if defined(USE_ASYNC_LDS)
      __builtin_amdgcn_s_wait_asynccnt(0);
#endif
    }
    __syncthreads();

    const float en = enorm[cb * 16 + l];
    v8f acc = {0.f, 0.f, 0.f, 0.f, 0.f, 0.f, 0.f, 0.f};
#pragma unroll
    for (int kc = 0; kc < 8; ++kc) {
      // B k order chosen here (same as A): lane (l,hi) holds column l, K = 16*hi + j
      AFrag bh, bl;
      const uint4* ph = (const uint4*)&sh_hi[cur][l * TPITCH + kc * 32 + hi * 16];
      const uint4* pl = (const uint4*)&sh_lo[cur][l * TPITCH + kc * 32 + hi * 16];
      bh.q[0] = ph[0]; bh.q[1] = ph[1];
      bl.q[0] = pl[0]; bl.q[1] = pl[1];
      acc = __builtin_amdgcn_wmma_f32_16x16x32_bf16(false, a_hi[kc].v, false, bh.v,
                                                    (short)0, acc, false, false);
      acc = __builtin_amdgcn_wmma_f32_16x16x32_bf16(false, a_hi[kc].v, false, bl.v,
                                                    (short)0, acc, false, false);
      acc = __builtin_amdgcn_wmma_f32_16x16x32_bf16(false, a_lo[kc].v, false, bh.v,
                                                    (short)0, acc, false, false);
    }
    const int code = cb * 16 + l;
#pragma unroll
    for (int r = 0; r < 8; ++r) {
      // score = ||e||^2 - 2 z.e  (||z||^2 constant per row: dropped for argmin)
      const float s = en - 2.0f * acc[r];
      if (s < minval[r]) { minval[r] = s; minidx[r] = code; }
    }
    __syncthreads();   // all waves done with buf[cur] before it is refilled
  }

  // C layout: lane (l,hi), vgpr r -> row r+8*hi, col l. Reduce over the 16 cols.
#pragma unroll
  for (int r = 0; r < 8; ++r) {
    float v = minval[r]; int i = minidx[r];
#pragma unroll
    for (int mask = 1; mask < 16; mask <<= 1) {
      const float ov = __shfl_xor(v, mask, 16);
      const int   oi = __shfl_xor(i, mask, 16);
      if (ov < v || (ov == v && oi < i)) { v = ov; i = oi; }  // first-min tiebreak
    }
    if (l == 0) {
      const int row = m0 + r + 8 * hi;
      idx_ws[row]  = i;
      idx_out[row] = (float)i;
    }
  }
}

// ---------------------------------------------------------------------------
// Kernel 3: gather z_q = emb[idx] back to NCHW (coalesced writes) + loss partials.
// One block per (b,h): 32 codebook rows staged in LDS, pitch 257 (conflict-free).
// ---------------------------------------------------------------------------
__global__ __launch_bounds__(256) void vq_gather(
    const float* __restrict__ z,
    const float* __restrict__ emb,
    const int* __restrict__ idx_ws,
    float* __restrict__ zq,
    float* __restrict__ partials)
{
  __shared__ int   s_idx[32];
  __shared__ float s_emb[32 * 257];
  __shared__ float s_red[8];
  const int t  = threadIdx.x;
  const int bh = blockIdx.x;                 // b*32 + h
  if (t < 32) s_idx[t] = idx_ws[bh * 32 + t];
  __syncthreads();
  {
    const int w = t >> 3, seg = t & 7;       // 8 threads per code row
    const float4* src = (const float4*)(emb + s_idx[w] * D_LAT + seg * 32);
#pragma unroll
    for (int q = 0; q < 8; ++q) {
      const float4 v = src[q];
      const int d = seg * 32 + q * 4;
      s_emb[w * 257 + d + 0] = v.x;
      s_emb[w * 257 + d + 1] = v.y;
      s_emb[w * 257 + d + 2] = v.z;
      s_emb[w * 257 + d + 3] = v.w;
    }
  }
  __syncthreads();
  const int w = t & 31, dq = t >> 5;
  const int b = bh >> 5, h = bh & 31;
  const int base = b * 262144 + h * 32 + w;  // contiguous in w across a wave
  float part = 0.f;
#pragma unroll 4
  for (int k = 0; k < 32; ++k) {
    const int d = dq + 8 * k;
    const float e  = s_emb[w * 257 + d];
    const int addr = base + d * 1024;
    const float zv = z[addr];
    zq[addr] = e;                            // straight-through forward value
    const float df = e - zv;
    part += df * df;
  }
#pragma unroll
  for (int mask = 1; mask < 32; mask <<= 1) part += __shfl_xor(part, mask, 32);
  if ((t & 31) == 0) s_red[t >> 5] = part;
  __syncthreads();
  if (t == 0) {
    float s = 0.f;
#pragma unroll
    for (int i = 0; i < 8; ++i) s += s_red[i];
    partials[bh] = s;                        // deterministic (no float atomics)
  }
}

// ---------------------------------------------------------------------------
// Kernel 4: reduce 1024 partials -> loss = (1+beta) * mean((z_q - z)^2)
// ---------------------------------------------------------------------------
__global__ __launch_bounds__(256) void vq_finalize(
    const float* __restrict__ partials, float* __restrict__ out_loss)
{
  __shared__ float s_red[8];
  const int t = threadIdx.x;
  float v = 0.f;
  for (int i = t; i < 1024; i += 256) v += partials[i];
#pragma unroll
  for (int mask = 1; mask < 32; mask <<= 1) v += __shfl_xor(v, mask, 32);
  if ((t & 31) == 0) s_red[t >> 5] = v;
  __syncthreads();
  if (t == 0) {
    float s = 0.f;
#pragma unroll
    for (int i = 0; i < 8; ++i) s += s_red[i];
    *out_loss = (1.0f + BETA) * s / (float)NUM_EL;
  }
}

extern "C" void kernel_launch(void* const* d_in, const int* in_sizes, int n_in,
                              void* d_out, int out_size, void* d_ws, size_t ws_size,
                              hipStream_t stream) {
  const float* z   = (const float*)d_in[0];   // [32,256,32,32] f32
  const float* emb = (const float*)d_in[1];   // [8192,256] f32

  char* ws = (char*)d_ws;
  unsigned short* ehi   = (unsigned short*)(ws);              // 4 MiB  bf16 hi
  unsigned short* elo   = (unsigned short*)(ws + 4194304);    // 4 MiB  bf16 lo
  float*          enorm = (float*)(ws + 8388608);             // 32 KiB ||e||^2
  int*            idxws = (int*)(ws + 8421376);               // 128 KiB idx
  float*          parts = (float*)(ws + 8552448);             // 4 KiB loss partials

  float* zq      = (float*)d_out;                 // [32,256,32,32]
  float* idx_out = (float*)d_out + NUM_EL;        // [32768] (as float)
  float* loss    = (float*)d_out + NUM_EL + N_ROWS;

  vq_prep    <<<K_CODES / 8, 256, 0, stream>>>(emb, ehi, elo, enorm);
  vq_argmin  <<<N_ROWS / 128, 256, 0, stream>>>(z, ehi, elo, enorm, idxws, idx_out);
  vq_gather  <<<1024,        256, 0, stream>>>(z, emb, idxws, zq, parts);
  vq_finalize<<<1,           256, 0, stream>>>(parts, loss);
}
